// HybridQNLPModel_17652315587525
// MI455X (gfx1250) — matrix operand order repaired and checked by reading, not compiled, not run on hardware
//
#include <hip/hip_runtime.h>
#include <hip/hip_bf16.h>

typedef _Float16 half_t;
typedef __attribute__((ext_vector_type(16))) _Float16 v16h;
typedef __attribute__((ext_vector_type(4)))  _Float16 h4;
typedef __attribute__((ext_vector_type(2)))  _Float16 h2;
typedef __attribute__((ext_vector_type(8)))  float    v8f;

#define N_QUBITS 8
#define DIMK     768
#define ROWS_PB  128   // batch rows per block (8 waves x 16 rows)
#define KCHUNK   128   // K staged per LDS chunk
#define XPITCH   132   // halves per LDS x row (66-word stride -> conflict-free)
#define WPITCH   770   // halves per LDS W row (385-word stride -> conflict-free)

__device__ __forceinline__ float shflx(float v, int m) { return __shfl_xor(v, m, 32); }
__device__ __forceinline__ float bcast(float v, int l) { return __shfl(v, l, 32); }

// State mapping: flat index s (qubit0 = MSB, matching reference reshape):
//   lane = s >> 3  (qubits 0..4 -> lane bit 4-q), j = s & 7 (qubits 5,6,7 -> j bit 7-q)
template<int Q>
__device__ __forceinline__ void ry_gate(float a[8], float c, float s, int lane) {
  if constexpr (Q <= 4) {
    constexpr int LM = 1 << (4 - Q);
    const bool hi = (lane & LM) != 0;
#pragma unroll
    for (int j = 0; j < 8; ++j) {
      float p = shflx(a[j], LM);                  // partner amplitude
      a[j] = hi ? fmaf(s, p, c * a[j])            // new1 = s*a0 + c*a1
                : fmaf(c, a[j], -s * p);          // new0 = c*a0 - s*a1
    }
  } else {
    constexpr int TB = 1 << (7 - Q);
#pragma unroll
    for (int j = 0; j < 8; ++j)
      if ((j & TB) == 0) {
        float a0 = a[j], a1 = a[j | TB];
        a[j]      = fmaf(c, a0, -s * a1);
        a[j | TB] = fmaf(s, a0,  c * a1);
      }
  }
}

template<int C, int T>
__device__ __forceinline__ void cnot_gate(float a[8], int lane) {
  if constexpr (C <= 4 && T <= 4) {               // both lane bits
    constexpr int CM = 1 << (4 - C), TM = 1 << (4 - T);
    const bool cc = (lane & CM) != 0;
#pragma unroll
    for (int j = 0; j < 8; ++j) {
      float v = shflx(a[j], TM);
      a[j] = cc ? v : a[j];
    }
  } else if constexpr (C <= 4 && T > 4) {         // ctrl lane bit, tgt in-register
    constexpr int CM = 1 << (4 - C), TB = 1 << (7 - T);
    const bool cc = (lane & CM) != 0;
#pragma unroll
    for (int j = 0; j < 8; ++j)
      if ((j & TB) == 0) {
        float a0 = a[j], a1 = a[j | TB];
        a[j]      = cc ? a1 : a0;
        a[j | TB] = cc ? a0 : a1;
      }
  } else if constexpr (C > 4 && T <= 4) {         // ctrl in-register, tgt lane bit
    constexpr int CB = 1 << (7 - C), TM = 1 << (4 - T);
#pragma unroll
    for (int j = 0; j < 8; ++j)
      if ((j & CB) != 0) a[j] = shflx(a[j], TM);  // uniform shuffle, uncond assign
  } else {                                        // both in-register
    constexpr int CB = 1 << (7 - C), TB = 1 << (7 - T);
#pragma unroll
    for (int j = 0; j < 8; ++j)
      if (((j & CB) != 0) && ((j & TB) == 0)) {
        float t = a[j]; a[j] = a[j | TB]; a[j | TB] = t;
      }
  }
}

__device__ __forceinline__ float wave_sum(float v) {
#pragma unroll
  for (int d = 1; d < 32; d <<= 1) v += shflx(v, d);
  return v;
}

__global__ __launch_bounds__(256)
void hqnlp_fused_kernel(const float* __restrict__ x, const float* __restrict__ Wm,
                        const float* __restrict__ scale, const float* __restrict__ qweights,
                        float* __restrict__ out) {
  __shared__ half_t sX[ROWS_PB][XPITCH];
  __shared__ half_t sW[16][WPITCH];   // rows 8..15 zero-padded: no masking in inner loop
  __shared__ float  sAng[8][16][8];   // [wave][row-in-tile][qubit]

  const int tid     = threadIdx.x;
  const int lane    = tid & 31;
  const int wave    = tid >> 5;
  const int rowBase = blockIdx.x * ROWS_PB;

  // ---- W (8x768 f32) -> LDS f16 once; pad rows 8..15 with zeros ----
  for (int r = 0; r < 16; ++r)
    for (int c = tid; c < DIMK; c += 256)
      sW[r][c] = (r < N_QUBITS) ? (half_t)Wm[r * DIMK + c] : (half_t)0.f;
  __syncthreads();

  const int h    = lane >> 4;   // half-wave
  const int m    = lane & 15;
  const int arow = wave * 16 + m;

  v8f acc = {0.f, 0.f, 0.f, 0.f, 0.f, 0.f, 0.f, 0.f};

  // ---- projection: stream x through LDS, WMMA-accumulate over K ----
  for (int kc = 0; kc < DIMK; kc += KCHUNK) {
#pragma unroll
    for (int it = 0; it < 16; ++it) {             // 128x128 f32 -> f16 LDS, coalesced
      int idx = it * 256 + tid;                   // float4 units (4096 total)
      int r   = idx >> 5;
      int c4  = idx & 31;
      const float4 v = *(const float4*)(x + (size_t)(rowBase + r) * DIMK + kc + c4 * 4);
      h4 hv = {(half_t)v.x, (half_t)v.y, (half_t)v.z, (half_t)v.w};
      *(h4*)&sX[r][c4 * 4] = hv;
    }
    if (kc + KCHUNK < DIMK)                       // hint next chunk (global_prefetch_b8)
      __builtin_prefetch(x + (size_t)(rowBase + (tid >> 1)) * DIMK + kc + KCHUNK + (tid & 1) * 64, 0, 1);
    __syncthreads();

#pragma unroll
    for (int ks = 0; ks < KCHUNK / 32; ++ks) {
      const int k0 = ks * 32;
      v16h a, b;
#pragma unroll
      for (int e = 0; e < 16; e += 2) {
        // A 16x32 f16 layout: K = e + 8*(e>=8) + 8*halfwave
        int ka = k0 + e + ((e & 8) ? 8 : 0) + 8 * h;
        h2 pa = *(const h2*)&sX[arow][ka];
        a[e] = pa.x; a[e + 1] = pa.y;
        // B 32x16 f16 layout (B = W^T, N>=8 rows are zero in LDS): K = e + 8*(e>=8) + 8*halfwave
        int kb = k0 + e + ((e & 8) ? 8 : 0) + 8 * h;
        h2 pb = *(const h2*)&sW[m][kc + kb];
        b[e] = pb.x; b[e + 1] = pb.y;
      }
      acc = __builtin_amdgcn_wmma_f32_16x16x32_f16(false, a, false, b, (short)0, acc,
                                                   false, false);
    }
    __syncthreads();
  }

  // ---- angles = tanh(proj)*scale -> LDS (D layout: acc[j] = D[m=j+8h][n=lane&15]) ----
  const float sc = scale[0];
  if (m < N_QUBITS) {
#pragma unroll
    for (int j = 0; j < 8; ++j)
      sAng[wave][j + 8 * h][m] = tanhf(acc[j]) * sc;
  }
  __syncthreads();

  // ---- per-lane variational-weight trig (2 layers x 8 qubits = 16 scalars) ----
  const float wq = qweights[lane & 15];
  const float cw = cosf(0.5f * wq);
  const float sw = sinf(0.5f * wq);

  // ---- one wave per batch element: 16 statevector sims per wave ----
  for (int e = 0; e < 16; ++e) {
    const float th = sAng[wave][e][lane & 7];
    const float ce = cosf(0.5f * th);
    const float se = sinf(0.5f * th);

    float a[8];
#pragma unroll
    for (int j = 0; j < 8; ++j) a[j] = 0.f;
    if (lane == 0) a[0] = 1.f;                    // |0...0>

    // encoding RYs
    ry_gate<0>(a, bcast(ce, 0), bcast(se, 0), lane);
    ry_gate<1>(a, bcast(ce, 1), bcast(se, 1), lane);
    ry_gate<2>(a, bcast(ce, 2), bcast(se, 2), lane);
    ry_gate<3>(a, bcast(ce, 3), bcast(se, 3), lane);
    ry_gate<4>(a, bcast(ce, 4), bcast(se, 4), lane);
    ry_gate<5>(a, bcast(ce, 5), bcast(se, 5), lane);
    ry_gate<6>(a, bcast(ce, 6), bcast(se, 6), lane);
    ry_gate<7>(a, bcast(ce, 7), bcast(se, 7), lane);

#pragma unroll
    for (int l = 0; l < 2; ++l) {
      const int o = l * 8;
      ry_gate<0>(a, bcast(cw, o + 0), bcast(sw, o + 0), lane);
      ry_gate<1>(a, bcast(cw, o + 1), bcast(sw, o + 1), lane);
      ry_gate<2>(a, bcast(cw, o + 2), bcast(sw, o + 2), lane);
      ry_gate<3>(a, bcast(cw, o + 3), bcast(sw, o + 3), lane);
      ry_gate<4>(a, bcast(cw, o + 4), bcast(sw, o + 4), lane);
      ry_gate<5>(a, bcast(cw, o + 5), bcast(sw, o + 5), lane);
      ry_gate<6>(a, bcast(cw, o + 6), bcast(sw, o + 6), lane);
      ry_gate<7>(a, bcast(cw, o + 7), bcast(sw, o + 7), lane);
      cnot_gate<0, 1>(a, lane);
      cnot_gate<1, 2>(a, lane);
      cnot_gate<2, 3>(a, lane);
      cnot_gate<3, 4>(a, lane);
      cnot_gate<4, 5>(a, lane);
      cnot_gate<5, 6>(a, lane);
      cnot_gate<6, 7>(a, lane);
      cnot_gate<7, 0>(a, lane);
    }

    // ---- Z expvals: sign(bit)*|amp|^2, reduced over the wave ----
    float ss = 0.f, p5 = 0.f, p6 = 0.f, p7 = 0.f;
#pragma unroll
    for (int j = 0; j < 8; ++j) {
      float v = a[j] * a[j];
      ss += v;
      p5 += (j & 4) ? -v : v;
      p6 += (j & 2) ? -v : v;
      p7 += (j & 1) ? -v : v;
    }
    float z0 = wave_sum((lane & 16) ? -ss : ss);
    float z1 = wave_sum((lane & 8)  ? -ss : ss);
    float z2 = wave_sum((lane & 4)  ? -ss : ss);
    float z3 = wave_sum((lane & 2)  ? -ss : ss);
    float z4 = wave_sum((lane & 1)  ? -ss : ss);
    float z5 = wave_sum(p5);
    float z6 = wave_sum(p6);
    float z7 = wave_sum(p7);

    if (lane == 0) {
      float* o = out + (size_t)(rowBase + wave * 16 + e) * N_QUBITS;
      o[0] = z0; o[1] = z1; o[2] = z2; o[3] = z3;
      o[4] = z4; o[5] = z5; o[6] = z6; o[7] = z7;
    }
  }
}

extern "C" void kernel_launch(void* const* d_in, const int* in_sizes, int n_in,
                              void* d_out, int out_size, void* d_ws, size_t ws_size,
                              hipStream_t stream) {
  (void)n_in; (void)out_size; (void)d_ws; (void)ws_size;
  const float* x     = (const float*)d_in[0];   // (65536, 768) f32
  const float* W     = (const float*)d_in[1];   // (8, 768) f32
  const float* scale = (const float*)d_in[2];   // (1,) f32
  const float* qw    = (const float*)d_in[3];   // (2, 8) f32
  float* out = (float*)d_out;                   // (65536, 8) f32

  const int B = in_sizes[0] / DIMK;             // 65536
  dim3 grid(B / ROWS_PB), block(256);
  hqnlp_fused_kernel<<<grid, block, 0, stream>>>(x, W, scale, qw, out);
}
